// IF_53111565582959
// MI455X (gfx1250) — matrix operand order repaired and checked by reading, not compiled, not run on hardware
//
#include <hip/hip_runtime.h>

// ---------------------------------------------------------------------------
// Problem constants
// ---------------------------------------------------------------------------
#define BB     2
#define CC     1280
#define HEADS  20
#define DH     64
#define INNER  1280
#define SS     4096
#define LL     77
#define ENC    4096
#define GROUPS 32
#define CPG    (CC / GROUPS)   // 40
#define EPSF   1e-5f

typedef __attribute__((ext_vector_type(16))) _Float16 v16h;
typedef __attribute__((ext_vector_type(8)))  _Float16 v8h;
typedef __attribute__((ext_vector_type(8)))  float    v8f;

__device__ inline v8f vzero8f() {
    v8f r;
#pragma unroll
    for (int i = 0; i < 8; ++i) r[i] = 0.f;
    return r;
}
__device__ inline v8h vzero8h() {
    v8h r;
#pragma unroll
    for (int i = 0; i < 8; ++i) r[i] = (_Float16)0.f;
    return r;
}

// A-matrix fragment (16x32 f16). Lane l: row = l&15, half h = l>>4.
// Element e<8  -> K = h*8 + e        (contiguous 8 halfs at base + h*8)
// Element e>=8 -> K = 16 + h*8 + e-8 (contiguous 8 halfs at base + 16 + h*8)
__device__ inline v16h load_a_frag(const _Float16* base, int h) {
    union { v16h v; v8h p[2]; } u;
    u.p[0] = *(const v8h*)(base + h * 8);
    u.p[1] = *(const v8h*)(base + 16 + h * 8);
    return u.v;
}
// B-matrix fragment (32x16 f16) from K-contiguous row (row index = N).
// Lane l: n = l&15, half h = l>>4; e<8 -> K = h*8 + e, e>=8 -> K = 16 + h*8 + e-8.
__device__ inline v16h load_b_frag(const _Float16* base, int h) {
    union { v16h v; v8h p[2]; } u;
    u.p[0] = *(const v8h*)(base + h * 8);
    u.p[1] = *(const v8h*)(base + 16 + h * 8);
    return u.v;
}
__device__ inline v8f wmma_f16(v16h a, v16h b, v8f c) {
    return __builtin_amdgcn_wmma_f32_16x16x32_f16(false, a, false, b,
                                                  (short)0, c, false, false);
}

// ---------------------------------------------------------------------------
// f32 -> f16 convert
// ---------------------------------------------------------------------------
__global__ void cvt_kernel(const float* __restrict__ in, _Float16* __restrict__ out, int n) {
    for (int i = blockIdx.x * blockDim.x + threadIdx.x; i < n; i += gridDim.x * blockDim.x)
        out[i] = (_Float16)in[i];
}

// ---------------------------------------------------------------------------
// GroupNorm statistics: one block per (b, g)
// ---------------------------------------------------------------------------
__global__ __launch_bounds__(256) void gn_stats_kernel(const float* __restrict__ x,
                                                       float* __restrict__ gnstat) {
    const int bg = blockIdx.x;             // 0..63
    const int b = bg >> 5, g = bg & 31;
    const float* base = x + ((size_t)b * CC + (size_t)g * CPG) * SS;
    float s = 0.f, ss = 0.f;
    for (int i = threadIdx.x; i < CPG * SS; i += 256) {
        float v = base[i];
        s += v; ss += v * v;
    }
    __shared__ float rs[256], rss[256];
    rs[threadIdx.x] = s; rss[threadIdx.x] = ss;
    __syncthreads();
    for (int st = 128; st > 0; st >>= 1) {
        if (threadIdx.x < st) {
            rs[threadIdx.x]  += rs[threadIdx.x + st];
            rss[threadIdx.x] += rss[threadIdx.x + st];
        }
        __syncthreads();
    }
    if (threadIdx.x == 0) {
        const float n  = (float)(CPG * SS);
        const float mu = rs[0] / n;
        const float var = rss[0] / n - mu * mu;
        gnstat[bg * 2 + 0] = mu;
        gnstat[bg * 2 + 1] = rsqrtf(var + EPSF);
    }
}

// Apply GN and write h transposed to [B, S, C] in f16. One block per (b, c).
__global__ __launch_bounds__(256) void gn_apply_kernel(const float* __restrict__ x,
                                                       const float* __restrict__ gnstat,
                                                       const float* __restrict__ gamma,
                                                       const float* __restrict__ beta,
                                                       _Float16* __restrict__ hf16) {
    const int bc = blockIdx.x;             // 0..2559
    const int b = bc / CC, c = bc % CC;
    const int g = c / CPG;
    const float mu   = gnstat[(b * GROUPS + g) * 2 + 0];
    const float rstd = gnstat[(b * GROUPS + g) * 2 + 1];
    const float ga = gamma[c] * rstd;
    const float be = beta[c] - mu * ga;
    const float* src = x + ((size_t)b * CC + c) * SS;
    _Float16* dst = hf16 + (size_t)b * SS * CC + c;
    for (int s = threadIdx.x; s < SS; s += 256)
        dst[(size_t)s * CC] = (_Float16)(src[s] * ga + be);
}

// ---------------------------------------------------------------------------
// LayerNorm on text: one block per row (B*L rows of ENC)
// ---------------------------------------------------------------------------
__global__ __launch_bounds__(256) void ln_kernel(const float* __restrict__ text,
                                                 const float* __restrict__ gamma,
                                                 const float* __restrict__ beta,
                                                 _Float16* __restrict__ enc16) {
    const int row = blockIdx.x;            // 0..153
    const float* src = text + (size_t)row * ENC;
    float s = 0.f, ss = 0.f;
    for (int i = threadIdx.x; i < ENC; i += 256) {
        float v = src[i];
        s += v; ss += v * v;
    }
    __shared__ float rs[256], rss[256];
    __shared__ float stat[2];
    rs[threadIdx.x] = s; rss[threadIdx.x] = ss;
    __syncthreads();
    for (int st = 128; st > 0; st >>= 1) {
        if (threadIdx.x < st) {
            rs[threadIdx.x]  += rs[threadIdx.x + st];
            rss[threadIdx.x] += rss[threadIdx.x + st];
        }
        __syncthreads();
    }
    if (threadIdx.x == 0) {
        const float n  = (float)ENC;
        const float mu = rs[0] / n;
        stat[0] = mu;
        stat[1] = rsqrtf(rss[0] / n - mu * mu + EPSF);
    }
    __syncthreads();
    const float mu = stat[0], rstd = stat[1];
    _Float16* dst = enc16 + (size_t)row * ENC;
    for (int i = threadIdx.x; i < ENC; i += 256)
        dst[i] = (_Float16)((src[i] - mu) * rstd * gamma[i] + beta[i]);
}

// ---------------------------------------------------------------------------
// Tiled WMMA GEMM: C[M,N] = A[M,K] @ Bw[K,N] + bias.
// Block tile 128x128, BK=32; 8 waves in 4x2 grid, each wave 32x64 (2x4 WMMA).
// EPI==0: store f16 [M,N].  EPI==1: out-proj epilogue -> transpose-store f32
//         d_out[b, n, s] = val + x[b, n, s]   (NOATTN=NORES=1)
// ---------------------------------------------------------------------------
template <int EPI>
__global__ __launch_bounds__(256) void gemm_kernel(const _Float16* __restrict__ A,
                                                   int M, int lda,
                                                   const _Float16* __restrict__ Bw,
                                                   int K, int N,
                                                   const float* __restrict__ bias,
                                                   _Float16* __restrict__ out16,
                                                   const float* __restrict__ xres,
                                                   float* __restrict__ outf) {
    constexpr int BM = 128, BN = 128, BK = 32, AS = 40;
    __shared__ _Float16 Asm[BM * AS];
    __shared__ _Float16 Bsm[BN * AS];

    const int tid  = threadIdx.x;
    const int lane = tid & 31, wid = tid >> 5;
    const int wm = wid >> 1, wn = wid & 1;
    const int m0 = blockIdx.y * BM, n0 = blockIdx.x * BN;
    const int ln15 = lane & 15, h = lane >> 4;

    v8f acc[2][4];
#pragma unroll
    for (int i = 0; i < 2; ++i)
#pragma unroll
        for (int j = 0; j < 4; ++j) acc[i][j] = vzero8f();

    for (int k0 = 0; k0 < K; k0 += BK) {
        // Stage A tile [128 rows][32 k], row-major, 16B vector loads.
        for (int c = tid; c < (BM * BK / 8); c += 256) {
            const int row = c >> 2, kc = c & 3;
            const int gr = m0 + row;
            v8h val = vzero8h();
            if (gr < M) val = *(const v8h*)(A + (size_t)gr * lda + k0 + kc * 8);
            *(v8h*)(&Asm[row * AS + kc * 8]) = val;
        }
        // Stage B tile transposed: Bsm[n][k] = Bw[k0+k][n0+n].
        {
            const int k = tid >> 3;            // 0..31
            const int n = (tid & 7) * 16;      // 0..112
            const _Float16* src = Bw + (size_t)(k0 + k) * N + n0 + n;
            v8h b0 = *(const v8h*)(src);
            v8h b1 = *(const v8h*)(src + 8);
#pragma unroll
            for (int j = 0; j < 8; ++j) Bsm[(n + j) * AS + k] = b0[j];
#pragma unroll
            for (int j = 0; j < 8; ++j) Bsm[(n + 8 + j) * AS + k] = b1[j];
        }
        __syncthreads();

        v16h af[2], bf[4];
#pragma unroll
        for (int i = 0; i < 2; ++i)
            af[i] = load_a_frag(&Asm[(wm * 32 + i * 16 + ln15) * AS], h);
#pragma unroll
        for (int j = 0; j < 4; ++j)
            bf[j] = load_b_frag(&Bsm[(wn * 64 + j * 16 + ln15) * AS], h);
#pragma unroll
        for (int i = 0; i < 2; ++i)
#pragma unroll
            for (int j = 0; j < 4; ++j)
                acc[i][j] = wmma_f16(af[i], bf[j], acc[i][j]);
        __syncthreads();
    }

    // Epilogue. C/D layout: lane -> N = ln15, rows r + 8*h (r = 0..7).
#pragma unroll
    for (int i = 0; i < 2; ++i) {
#pragma unroll
        for (int j = 0; j < 4; ++j) {
            const int n = n0 + wn * 64 + j * 16 + ln15;
            const float bv = bias ? bias[n] : 0.f;
#pragma unroll
            for (int r = 0; r < 8; ++r) {
                const int m = m0 + wm * 32 + i * 16 + r + 8 * h;
                if (m >= M) continue;
                const float v = acc[i][j][r] + bv;
                if (EPI == 0) {
                    out16[(size_t)m * N + n] = (_Float16)v;
                } else {
                    const int b = m >> 12;       // m / S
                    const int s = m & (SS - 1);  // m % S
                    const size_t idx = ((size_t)b * CC + n) * SS + s;
                    outf[idx] = v + xres[idx];
                }
            }
        }
    }
}

// ---------------------------------------------------------------------------
// Attention: grid (S/128, B*HEADS), 256 threads (8 waves x 16 rows = 128 rows).
// sim = (q @ K^T) * scale  -> conditional-half mask regularizer -> softmax
// out = probs @ V, written f16 to attn16[b*S + s][head*64 + dh].
// ---------------------------------------------------------------------------
__global__ __launch_bounds__(256) void attn_kernel(const _Float16* __restrict__ q16,
                                                   const _Float16* __restrict__ k16,
                                                   const _Float16* __restrict__ v16,
                                                   const float* __restrict__ size_reg,
                                                   const float* __restrict__ tregp,
                                                   const int* __restrict__ mask,
                                                   _Float16* __restrict__ attn16) {
    constexpr int KT = 72;    // Klds row stride (halfs)
    constexpr int VT = 104;   // Vt row stride
    constexpr int PS = 96;    // probs row stride == padded K of 2nd matmul
    __shared__ _Float16 Klds[80 * KT];      // [l][dh], l padded to 80
    __shared__ _Float16 Vt[DH * VT];        // [dh][l], l padded to 96
    __shared__ _Float16 probsL[8 * 16 * PS];

    const int tid  = threadIdx.x;
    const int lane = tid & 31, wid = tid >> 5;
    const int ln15 = lane & 15, h = lane >> 4;
    const int bh = blockIdx.y;
    const int b = bh / HEADS, head = bh % HEADS;
    const int s0 = blockIdx.x * 128 + wid * 16;

    // Zero this wave's probs region (covers padded columns 77..95).
    for (int i = lane; i < 16 * PS; i += 32)
        probsL[wid * 16 * PS + i] = (_Float16)0.f;

    // Stage K: Klds[l][dh]
    for (int i = tid; i < 80 * 64; i += 256) {
        const int n = i >> 6, k = i & 63;
        _Float16 val = (_Float16)0.f;
        if (n < LL) val = k16[((size_t)(b * LL + n)) * INNER + head * DH + k];
        Klds[n * KT + k] = val;
    }
    // Stage V transposed: Vt[dh][l]
    for (int i = tid; i < DH * PS; i += 256) {
        const int d = i / PS, l = i % PS;
        _Float16 val = (_Float16)0.f;
        if (l < LL) val = v16[((size_t)(b * LL + l)) * INNER + head * DH + d];
        Vt[d * VT + l] = val;
    }
    __syncthreads();

    // ---- sim = q @ K^T : 5 N-tiles (cols 0..79), K = DH = 64 (2 steps) ----
    v8f sacc[5];
#pragma unroll
    for (int t = 0; t < 5; ++t) sacc[t] = vzero8f();

    const _Float16* qrow =
        q16 + ((size_t)(b * SS + s0 + ln15)) * INNER + head * DH;
#pragma unroll
    for (int ks = 0; ks < 2; ++ks) {
        v16h af = load_a_frag(qrow + ks * 32, h);
#pragma unroll
        for (int t = 0; t < 5; ++t) {
            v16h bf = load_b_frag(&Klds[(t * 16 + ln15) * KT + ks * 32], h);
            sacc[t] = wmma_f16(af, bf, sacc[t]);
        }
    }

    // ---- mask regularizer + softmax (rows r + 8h, cols 16t + ln15) ----
    const float tw = tregp[0];           // CREG == 1.0
    const float scale = 0.125f;          // DH^-0.5
#pragma unroll
    for (int r = 0; r < 8; ++r) {
        const int row = r + 8 * h;
        const int s = s0 + row;
        float v[5];
        bool ok[5];
        float mn = 3.4e38f, mx = -3.4e38f;
#pragma unroll
        for (int t = 0; t < 5; ++t) {
            const int col = t * 16 + ln15;
            ok[t] = col < LL;
            v[t] = sacc[t][r] * scale;
            if (ok[t]) { mn = fminf(mn, v[t]); mx = fmaxf(mx, v[t]); }
        }
#pragma unroll
        for (int off = 8; off > 0; off >>= 1) {
            mn = fminf(mn, __shfl_xor(mn, off, 16));
            mx = fmaxf(mx, __shfl_xor(mx, off, 16));
        }
        if (b == 1) {  // conditional half
#pragma unroll
            for (int t = 0; t < 5; ++t) {
                if (!ok[t]) continue;
                const int col = t * 16 + ln15;
                const size_t mi = (size_t)s * LL + col;
                const float w = size_reg[mi] * tw;
                if (mask[mi] > 0) v[t] += w * (mx - v[t]);
                else              v[t] -= w * (v[t] - mn);
            }
            mx = -3.4e38f;
#pragma unroll
            for (int t = 0; t < 5; ++t)
                if (ok[t]) mx = fmaxf(mx, v[t]);
#pragma unroll
            for (int off = 8; off > 0; off >>= 1)
                mx = fmaxf(mx, __shfl_xor(mx, off, 16));
        }
        float e[5];
        float sum = 0.f;
#pragma unroll
        for (int t = 0; t < 5; ++t) {
            e[t] = ok[t] ? __expf(v[t] - mx) : 0.f;
            sum += e[t];
        }
#pragma unroll
        for (int off = 8; off > 0; off >>= 1)
            sum += __shfl_xor(sum, off, 16);
        const float inv = 1.f / sum;
#pragma unroll
        for (int t = 0; t < 5; ++t) {
            const int col = t * 16 + ln15;
            if (col < 80)
                probsL[wid * 16 * PS + row * PS + col] =
                    ok[t] ? (_Float16)(e[t] * inv) : (_Float16)0.f;
        }
    }

    // LDS ops from the same wave complete in order; fence before re-reading.
    asm volatile("s_wait_dscnt 0" ::: "memory");

    // ---- out = probs @ V : 4 N-tiles (dh), K = 96 (3 steps) ----
    v8f oacc[4];
#pragma unroll
    for (int j = 0; j < 4; ++j) oacc[j] = vzero8f();
    const _Float16* prow = &probsL[wid * 16 * PS + ln15 * PS];
#pragma unroll
    for (int ks = 0; ks < 3; ++ks) {
        v16h af = load_a_frag(prow + ks * 32, h);
#pragma unroll
        for (int j = 0; j < 4; ++j) {
            v16h bf = load_b_frag(&Vt[(j * 16 + ln15) * VT + ks * 32], h);
            oacc[j] = wmma_f16(af, bf, oacc[j]);
        }
    }
#pragma unroll
    for (int j = 0; j < 4; ++j) {
        const int n = head * DH + j * 16 + ln15;
#pragma unroll
        for (int r = 0; r < 8; ++r) {
            const int s = s0 + r + 8 * h;
            attn16[((size_t)(b * SS + s)) * INNER + n] = (_Float16)oacc[j][r];
        }
    }
}

// ---------------------------------------------------------------------------
// Host launch
// ---------------------------------------------------------------------------
extern "C" void kernel_launch(void* const* d_in, const int* in_sizes, int n_in,
                              void* d_out, int out_size, void* d_ws, size_t ws_size,
                              hipStream_t stream) {
    const float* x        = (const float*)d_in[0];
    const float* text     = (const float*)d_in[1];
    const float* gn_gamma = (const float*)d_in[2];
    const float* gn_beta  = (const float*)d_in[3];
    const float* ln_gamma = (const float*)d_in[4];
    const float* ln_beta  = (const float*)d_in[5];
    const float* Wq       = (const float*)d_in[6];
    const float* bq       = (const float*)d_in[7];
    const float* Wk       = (const float*)d_in[8];
    const float* bk       = (const float*)d_in[9];
    const float* Wv       = (const float*)d_in[10];
    const float* bv       = (const float*)d_in[11];
    const float* Wo       = (const float*)d_in[12];
    const float* bo       = (const float*)d_in[13];
    const float* size_reg = (const float*)d_in[14];
    const float* treg     = (const float*)d_in[15];
    const int*   mask     = (const int*)d_in[16];
    float* out = (float*)d_out;

    char* p = (char*)d_ws;
    auto alloc = [&](size_t bytes) -> char* {
        char* r = p;
        p += (bytes + 255) & ~(size_t)255;
        return r;
    };
    float*     gnstat = (float*)alloc((size_t)BB * GROUPS * 2 * sizeof(float));
    _Float16*  hf16   = (_Float16*)alloc((size_t)BB * SS * CC * 2);
    _Float16*  enc16  = (_Float16*)alloc((size_t)BB * LL * ENC * 2);
    _Float16*  Wq16   = (_Float16*)alloc((size_t)CC * INNER * 2);
    _Float16*  Wk16   = (_Float16*)alloc((size_t)ENC * INNER * 2);
    _Float16*  Wv16   = (_Float16*)alloc((size_t)ENC * INNER * 2);
    _Float16*  Wo16   = (_Float16*)alloc((size_t)INNER * CC * 2);
    _Float16*  q16    = (_Float16*)alloc((size_t)BB * SS * INNER * 2);
    _Float16*  k16    = (_Float16*)alloc((size_t)160 * INNER * 2);
    _Float16*  v16w   = (_Float16*)alloc((size_t)160 * INNER * 2);
    _Float16*  attn16 = (_Float16*)alloc((size_t)BB * SS * INNER * 2);

    cvt_kernel<<<512, 256, 0, stream>>>(Wq, Wq16, CC * INNER);
    cvt_kernel<<<512, 256, 0, stream>>>(Wk, Wk16, ENC * INNER);
    cvt_kernel<<<512, 256, 0, stream>>>(Wv, Wv16, ENC * INNER);
    cvt_kernel<<<512, 256, 0, stream>>>(Wo, Wo16, INNER * CC);

    gn_stats_kernel<<<BB * GROUPS, 256, 0, stream>>>(x, gnstat);
    gn_apply_kernel<<<BB * CC, 256, 0, stream>>>(x, gnstat, gn_gamma, gn_beta, hf16);
    ln_kernel<<<BB * LL, 256, 0, stream>>>(text, ln_gamma, ln_beta, enc16);

    // q = h @ Wq + bq            M=8192 K=1280 N=1280
    gemm_kernel<0><<<dim3(INNER / 128, (BB * SS) / 128), 256, 0, stream>>>(
        hf16, BB * SS, CC, Wq16, CC, INNER, bq, q16, nullptr, nullptr);
    // k = enc @ Wk + bk          M=154 K=4096 N=1280
    gemm_kernel<0><<<dim3(INNER / 128, 2), 256, 0, stream>>>(
        enc16, BB * LL, ENC, Wk16, ENC, INNER, bk, k16, nullptr, nullptr);
    // v = enc @ Wv + bv
    gemm_kernel<0><<<dim3(INNER / 128, 2), 256, 0, stream>>>(
        enc16, BB * LL, ENC, Wv16, ENC, INNER, bv, v16w, nullptr, nullptr);

    // attention
    attn_kernel<<<dim3(SS / 128, BB * HEADS), 256, 0, stream>>>(
        q16, k16, v16w, size_reg, treg, mask, attn16);

    // out = attn @ Wo + bo, transposed store + residual add
    gemm_kernel<1><<<dim3(CC / 128, (BB * SS) / 128), 256, 0, stream>>>(
        attn16, BB * SS, INNER, Wo16, INNER, CC, bo, nullptr, x, out);
}
